// GraphsageMEAN_80023830659316
// MI455X (gfx1250) — matrix operand order repaired and checked by reading, not compiled, not run on hardware
//
#include <hip/hip_runtime.h>
#include <hip/hip_bf16.h>

typedef __bf16 bf16_t;
typedef __attribute__((ext_vector_type(16))) __bf16 v16bf;
typedef __attribute__((ext_vector_type(8)))  float  v8f;

#define NN 100000
#define NE 1600000

// ---------------- CSR build: histogram of dst ----------------
__global__ void k_hist(const long long* __restrict__ dst, int* __restrict__ cnt, int E) {
    int t = blockIdx.x * blockDim.x + threadIdx.x;
    if (t < E) atomicAdd(&cnt[(int)dst[t]], 1);
}

// Single-block exclusive scan: cnt[0..N) -> rowptr[0..N]
__global__ __launch_bounds__(1024) void k_scan_rowptr(const int* __restrict__ cnt,
                                                      int* __restrict__ rowptr, int N) {
    __shared__ int sh[1024];
    const int T   = 1024;
    const int t   = threadIdx.x;
    const int per = (N + T - 1) / T;
    const int lo  = t * per;
    const int hi  = min(lo + per, N);
    int s = 0;
    for (int i = lo; i < hi; ++i) s += cnt[i];
    sh[t] = s;
    __syncthreads();
    for (int d = 1; d < T; d <<= 1) {           // inclusive Hillis-Steele
        int v = (t >= d) ? sh[t - d] : 0;
        __syncthreads();
        sh[t] += v;
        __syncthreads();
    }
    int run = (t > 0) ? sh[t - 1] : 0;          // exclusive base for this segment
    for (int i = lo; i < hi; ++i) { rowptr[i] = run; run += cnt[i]; }
    if (t == T - 1) rowptr[N] = run;            // grand total
}

// Fill eidx with source node ids, bucketed by dst
__global__ void k_csr_fill(const long long* __restrict__ src, const long long* __restrict__ dst,
                           int* __restrict__ off, int* __restrict__ eidx, int E) {
    int t = blockIdx.x * blockDim.x + threadIdx.x;
    if (t < E) {
        int d = (int)dst[t];
        int p = atomicAdd(&off[d], 1);
        eidx[p] = (int)src[t];
    }
}

__global__ void k_dinv(const int* __restrict__ cnt, float* __restrict__ dinv, int N) {
    int t = blockIdx.x * blockDim.x + threadIdx.x;
    if (t < N) dinv[t] = 1.0f / fmaxf((float)cnt[t], 1.0f);
}

// ---------------- f32 -> bf16 weight conversion (tiny, once per call) ----------------
__global__ void k_cvt_bf16(const float* __restrict__ w, bf16_t* __restrict__ o, int n) {
    int t = blockIdx.x * blockDim.x + threadIdx.x;
    if (t < n) o[t] = (bf16_t)w[t];
}

// ---------------- CSR gather aggregation: agg[n] = sum_{s in nbrs(n)} h[s] ----------------
// FV = F/4 lanes cooperate on one node; pure gather + coalesced store, zero f32 atomics.
template<int F>
__global__ __launch_bounds__(256) void k_aggregate_csr(
    const float* __restrict__ h, const int* __restrict__ rowptr,
    const int* __restrict__ eidx, float* __restrict__ agg, int N) {
    const int FV   = F / 4;
    const int gtid = blockIdx.x * blockDim.x + threadIdx.x;
    const int node = gtid / FV;
    const int c    = gtid & (FV - 1);
    if (node >= N) return;
    const int beg = rowptr[node];
    const int end = rowptr[node + 1];
    float4 acc = {0.f, 0.f, 0.f, 0.f};
    for (int j = beg; j < end; ++j) {
        const int s = eidx[j];                       // same addr across group -> 1 req/wave
        const float4 v = ((const float4*)(h + (size_t)s * F))[c];
        acc.x += v.x; acc.y += v.y; acc.z += v.z; acc.w += v.w;
    }
    ((float4*)(agg + (size_t)node * F))[c] = acc;    // full overwrite, no memset needed
}

// Fill 8 consecutive elements of a bf16 A-operand fragment from 8 contiguous f32, scaled.
__device__ __forceinline__ void a_fill8(v16bf& a, int base, const float* __restrict__ p, float s) {
    float4 f0 = *(const float4*)(p);
    float4 f1 = *(const float4*)(p + 4);
    a[base + 0] = (bf16_t)(f0.x * s); a[base + 1] = (bf16_t)(f0.y * s);
    a[base + 2] = (bf16_t)(f0.z * s); a[base + 3] = (bf16_t)(f0.w * s);
    a[base + 4] = (bf16_t)(f1.x * s); a[base + 5] = (bf16_t)(f1.y * s);
    a[base + 6] = (bf16_t)(f1.z * s); a[base + 7] = (bf16_t)(f1.w * s);
}

// ---------------- node update: out = mean*Wl^T + h*Wr^T + b (one wave per 16-node tile) ----
// A (16-bit 16x32): lane L holds row M=L&15; lanes 0-15 K 0..15, lanes 16-31 K 16..31.
// B (32x16): lane L holds col N=L&15; 16 contiguous K per lane -> one 32B row-major load.
// C/D: VGPR v, lane L: M = (L>>4)*8 + v, N = L&15.
template<int FIN, int FOUT, bool RELU>
__global__ __launch_bounds__(256) void k_sage_update(
    const float* __restrict__ h, const float* __restrict__ agg,
    const float* __restrict__ dinv,
    const bf16_t* __restrict__ Wl, const bf16_t* __restrict__ Wr,
    const float* __restrict__ bias, float* __restrict__ out, int nTiles) {

    const int lane = threadIdx.x & 31;
    const int wave = threadIdx.x >> 5;
    const int tile = blockIdx.x * (blockDim.x >> 5) + wave;
    if (tile >= nTiles) return;

    const int row  = lane & 15;
    const int half = lane >> 4;
    const int node = tile * 16 + row;           // 100000 is an exact multiple of 16
    const float s  = dinv[node];
    const float* __restrict__ hrow = h   + (size_t)node * FIN;
    const float* __restrict__ arow = agg + (size_t)node * FIN;

    const int ncol = lane & 15;
    const int koff = half * 16;

    #pragma unroll
    for (int ct = 0; ct < FOUT / 16; ++ct) {
        v8f acc = {};
        const bf16_t* __restrict__ wl = Wl + (size_t)(ct * 16 + ncol) * FIN + half * 16;
        const bf16_t* __restrict__ wr = Wr + (size_t)(ct * 16 + ncol) * FIN + half * 16;
        #pragma unroll
        for (int kk = 0; kk < FIN / 32; ++kk) {
            const int kb = kk * 32;
            v16bf Am, Ah, Bl, Br;
            a_fill8(Am, 0, arow + kb + koff,      s);
            a_fill8(Am, 8, arow + kb + 8 + koff, s);
            a_fill8(Ah, 0, hrow + kb + koff,      1.0f);
            a_fill8(Ah, 8, hrow + kb + 8 + koff, 1.0f);
            Bl = *(const v16bf*)(wl + kb);
            Br = *(const v16bf*)(wr + kb);
            acc = __builtin_amdgcn_wmma_f32_16x16x32_bf16(false, Am, false, Bl,
                                                          (short)0, acc, false, false);
            acc = __builtin_amdgcn_wmma_f32_16x16x32_bf16(false, Ah, false, Br,
                                                          (short)0, acc, false, false);
        }
        const int col = ct * 16 + ncol;
        const float bv = bias[col];
        #pragma unroll
        for (int v = 0; v < 8; ++v) {
            const int m = half * 8 + v;
            float val = acc[v] + bv;
            if (RELU) val = fmaxf(val, 0.0f);
            out[(size_t)(tile * 16 + m) * FOUT + col] = val;
        }
    }
}

extern "C" void kernel_launch(void* const* d_in, const int* in_sizes, int n_in,
                              void* d_out, int out_size, void* d_ws, size_t ws_size,
                              hipStream_t stream) {
    const float*     x   = (const float*)d_in[0];
    const long long* ei  = (const long long*)d_in[1];
    const long long* src = ei;
    const long long* dst = ei + NE;
    const float* Wl1 = (const float*)d_in[2];
    const float* Wr1 = (const float*)d_in[3];
    const float* b1  = (const float*)d_in[4];
    const float* Wl2 = (const float*)d_in[5];
    const float* Wr2 = (const float*)d_in[6];
    const float* b2  = (const float*)d_in[7];
    const float* Wl3 = (const float*)d_in[8];
    const float* Wr3 = (const float*)d_in[9];
    const float* b3  = (const float*)d_in[10];
    float* out = (float*)d_out;   // also used as storage for h1 (exactly N*64)

    // ---- workspace carve-up (~111 MB) ----
    float*  agg    = (float*)d_ws;                       // N*128 f32
    float*  h2     = agg + (size_t)NN * 128;             // N*128 f32
    float*  dinv   = h2  + (size_t)NN * 128;             // N f32
    int*    cnt    = (int*)(dinv + NN);                  // N i32
    int*    rowptr = cnt + NN;                           // N+1 i32
    int*    off    = rowptr + NN + 1;                    // N i32
    int*    eidx   = off + NN + 63;                      // NE i32 (padded for alignment)
    bf16_t* wl1b   = (bf16_t*)(eidx + NE);               // 6 x 8192 bf16
    bf16_t* wr1b   = wl1b + 8192;
    bf16_t* wl2b   = wr1b + 8192;
    bf16_t* wr2b   = wl2b + 8192;
    bf16_t* wl3b   = wr2b + 8192;
    bf16_t* wr3b   = wl3b + 8192;

    const int TPB   = 256;
    const int tiles = NN / 16;                 // 6250 (exact)
    const int ugrid = (tiles + 7) / 8;         // 8 waves/block

    // ---- CSR build (once; reused by all 3 layers) ----
    hipMemsetAsync(cnt, 0, (size_t)NN * sizeof(int), stream);
    k_hist<<<(NE + TPB - 1) / TPB, TPB, 0, stream>>>(dst, cnt, NE);
    k_scan_rowptr<<<1, 1024, 0, stream>>>(cnt, rowptr, NN);
    hipMemcpyAsync(off, rowptr, (size_t)NN * sizeof(int), hipMemcpyDeviceToDevice, stream);
    k_csr_fill<<<(NE + TPB - 1) / TPB, TPB, 0, stream>>>(src, dst, off, eidx, NE);
    k_dinv<<<(NN + TPB - 1) / TPB, TPB, 0, stream>>>(cnt, dinv, NN);

    // ---- weight conversion (tiny) ----
    k_cvt_bf16<<<32, TPB, 0, stream>>>(Wl1, wl1b, 8192);
    k_cvt_bf16<<<32, TPB, 0, stream>>>(Wr1, wr1b, 8192);
    k_cvt_bf16<<<32, TPB, 0, stream>>>(Wl2, wl2b, 8192);
    k_cvt_bf16<<<32, TPB, 0, stream>>>(Wr2, wr2b, 8192);
    k_cvt_bf16<<<32, TPB, 0, stream>>>(Wl3, wl3b, 8192);
    k_cvt_bf16<<<32, TPB, 0, stream>>>(Wr3, wr3b, 8192);

    // ---- layer 1: 128 -> 64, input x, output h1 (stored in d_out) ----
    k_aggregate_csr<128><<<(NN * 32 + TPB - 1) / TPB, TPB, 0, stream>>>(x, rowptr, eidx, agg, NN);
    k_sage_update<128, 64, true><<<ugrid, TPB, 0, stream>>>(x, agg, dinv, wl1b, wr1b, b1, out, tiles);

    // ---- layer 2: 64 -> 128, input h1 (=out), output h2 ----
    k_aggregate_csr<64><<<(NN * 16 + TPB - 1) / TPB, TPB, 0, stream>>>(out, rowptr, eidx, agg, NN);
    k_sage_update<64, 128, true><<<ugrid, TPB, 0, stream>>>(out, agg, dinv, wl2b, wr2b, b2, h2, tiles);

    // ---- layer 3: 128 -> 64, input h2, output d_out ----
    k_aggregate_csr<128><<<(NN * 32 + TPB - 1) / TPB, TPB, 0, stream>>>(h2, rowptr, eidx, agg, NN);
    k_sage_update<128, 64, false><<<ugrid, TPB, 0, stream>>>(h2, agg, dinv, wl3b, wr3b, b3, out, tiles);
}
